// CrossNet_68470368633663
// MI455X (gfx1250) — matrix operand order repaired and checked by reading, not compiled, not run on hardware
//
#include <hip/hip_runtime.h>
#include <hip/hip_bf16.h>
#include <math.h>

// CrossNet MoE on MI455X (gfx1250, wave32).
// Per layer: gate (noisy top-2 softmax) then a single fused GEMM
//   y[b,o] = sum_{e,k} (g[b,e]*x[b,k]) * W[l,e,o,k]
// with epilogue x_next = x0 * (y + sum_e g[b,e]*bias[l,e,o]) + x.
// f32 precision recovered via bf16 hi/lo split (3 WMMA products per tile).

#define LAYERS 4
#define NEXP   8
#define D      512
#define GH     256
#define BTOK   4096

#define BM 64
#define BN 64
#define BK 64
#define LDA 72   // ushort row stride (pad 8): 144B rows, 16B aligned, conflict-free

typedef __attribute__((ext_vector_type(16))) __bf16 v16bf;
typedef __attribute__((ext_vector_type(8)))  float  v8f;

union Frag { v16bf v; uint4 q[2]; };

struct SharedTiles {
  alignas(16) unsigned short Ahi[BM][LDA];
  alignas(16) unsigned short Alo[BM][LDA];
  alignas(16) unsigned short Bhi[BN][LDA];
  alignas(16) unsigned short Blo[BN][LDA];
};

__device__ __forceinline__ unsigned pcg_hash(unsigned v) {
  v = v * 747796405u + 2891336453u;
  unsigned w = ((v >> ((v >> 28u) + 4u)) ^ v) * 277803737u;
  return (w >> 22u) ^ w;
}

// Round-to-nearest-even f32 -> (bf16 hi, bf16 lo) split: x ~= hi + lo.
__device__ __forceinline__ void split_f32_bf16(float x, unsigned short& hi,
                                               unsigned short& lo) {
  union { float f; unsigned u; } a; a.f = x;
  unsigned r = a.u + 0x7FFFu + ((a.u >> 16) & 1u);
  hi = (unsigned short)(r >> 16);
  union { unsigned u; float f; } h; h.u = ((unsigned)hi) << 16;
  union { float f; unsigned u; } b; b.f = x - h.f;
  unsigned r2 = b.u + 0x7FFFu + ((b.u >> 16) & 1u);
  lo = (unsigned short)(r2 >> 16);
}

// ---------------------------------------------------------------------------
// Gate kernel: one wave32 per token. h = relu(x@Wg1^T); logits = h@Wg2^T;
// sp = softplus(x@Wn^T); s = logits + eps*sp (hash-based normal stands in for
// jax threefry — bit-exact replication of JAX RNG is not feasible on-device);
// top-2 threshold mask; softmax -> gate[b, 0..7].
// ---------------------------------------------------------------------------
__global__ __launch_bounds__(128) void gate_kernel(
    const float* __restrict__ x, const float* __restrict__ Wg1,
    const float* __restrict__ Wg2, const float* __restrict__ Wn,
    float* __restrict__ gate, int layer) {
  __shared__ float sx[4][D];
  const int tid = threadIdx.x;
  const int b0 = blockIdx.x * 4;

  // stage 4 tokens of x into LDS (512 float4 total, 4 per thread)
  const float4* xg = (const float4*)(x + (size_t)b0 * D);
  float4* sx4 = (float4*)&sx[0][0];
#pragma unroll
  for (int i = 0; i < 4; ++i) { int idx = tid + 128 * i; sx4[idx] = xg[idx]; }
  __syncthreads();

  const int wave = tid >> 5, lane = tid & 31;
  const int b = b0 + wave;
  const float* xv = sx[wave];

  // hidden: each lane owns 8 of 256 dims
  float hl[8];
#pragma unroll
  for (int i = 0; i < 8; ++i) {
    const int j = i * 32 + lane;
    const float4* wr = (const float4*)(Wg1 + (size_t)j * D);
    const float4* xr = (const float4*)xv;
    float acc = 0.f;
    for (int k = 0; k < D / 4; ++k) {
      float4 a = xr[k], w = wr[k];
      acc += a.x * w.x + a.y * w.y + a.z * w.z + a.w * w.w;
    }
    hl[i] = fmaxf(acc, 0.f);
  }

  float s[8];
#pragma unroll
  for (int e = 0; e < NEXP; ++e) {
    float p = 0.f;
#pragma unroll
    for (int i = 0; i < 8; ++i) p += hl[i] * Wg2[e * GH + i * 32 + lane];
#pragma unroll
    for (int off = 16; off > 0; off >>= 1) p += __shfl_xor(p, off, 32);

    float q = 0.f;
#pragma unroll
    for (int i = 0; i < 16; ++i) { int k = i * 32 + lane; q += xv[k] * Wn[e * D + k]; }
#pragma unroll
    for (int off = 16; off > 0; off >>= 1) q += __shfl_xor(q, off, 32);
    float sp = (q > 20.f) ? q : log1pf(__expf(q));  // softplus

    unsigned seed = (unsigned)(b * NEXP + e) ^ (0x9E3779B9u * (unsigned)(layer + 1)) ^ 42u;
    unsigned r1 = pcg_hash(seed), r2 = pcg_hash(seed ^ 0x85ebca6bu);
    float u1 = ((float)r1 + 0.5f) * 2.3283064365386963e-10f;
    float u2 = ((float)r2 + 0.5f) * 2.3283064365386963e-10f;
    float eps = sqrtf(-2.f * __logf(u1)) * __cosf(6.283185307179586f * u2);
    s[e] = p + eps * sp;
  }

  // top-2 threshold (2nd largest), masked softmax
  float m1 = -1e30f, m2 = -1e30f;
#pragma unroll
  for (int e = 0; e < NEXP; ++e) {
    float v = s[e];
    if (v > m1) { m2 = m1; m1 = v; } else if (v > m2) { m2 = v; }
  }
  float pr[8]; float sum = 0.f;
#pragma unroll
  for (int e = 0; e < NEXP; ++e) {
    float v = (s[e] < m2) ? 0.f : __expf(s[e] - m1);
    pr[e] = v; sum += v;
  }
  const float inv = 1.f / sum;
  if (lane == 0) {
#pragma unroll
    for (int e = 0; e < NEXP; ++e) gate[(size_t)b * NEXP + e] = pr[e] * inv;
  }
}

// ---------------------------------------------------------------------------
// Fused MoE layer: y = A'(4096x4096) @ W'(4096x512), A'[b,(e,k)] = g[b,e]*x[b,k],
// W'[(e,k),o] = W[l,e,o,k]. 64x64 C tile per 256-thread block (8 wave32s, 2
// 16x16 subtiles each). bf16 hi/lo split -> 3x v_wmma_f32_16x16x32_bf16.
// Epilogue fuses bias, gate, x0 multiply and residual.
// ---------------------------------------------------------------------------
__global__ __launch_bounds__(256) void moe_layer_kernel(
    const float* __restrict__ xin, const float* __restrict__ x0,
    const float* __restrict__ Wl, const float* __restrict__ bl,
    const float* __restrict__ gate, float* __restrict__ xout) {
  __shared__ SharedTiles sm;
  const int tid = threadIdx.x;
  const int lane = tid & 31, wave = tid >> 5;
  const int half = lane >> 4, lr = lane & 15;
  const int bN = blockIdx.x * BN;  // output-dim block
  const int bM = blockIdx.y * BM;  // token block
  const int m_off = (wave & 3) * 16;
  const int n0 = (wave >> 2) * 32;
  const int n1 = n0 + 16;

  v8f acc0 = {}; v8f acc1 = {};

  const int row = tid & 63;   // staging row (token for A, out-dim for B)
  const int c4b = tid >> 6;   // 0..3 -> float4 column groups

  const int NKT = (NEXP * D) / BK;  // 64 k-tiles
  for (int kt = 0; kt < NKT; ++kt) {
    const int gk = kt * BK;
    const int e  = gk >> 9;        // expert slab (512 k each)
    const int ke = gk & (D - 1);   // k offset within expert

    // prefetch next W tile row (hits L2; W layer is fully L2-resident)
    if (kt + 1 < NKT) {
      const int gk2 = gk + BK;
      __builtin_prefetch(Wl + (((size_t)(gk2 >> 9)) * D + (bN + row)) * D + (gk2 & (D - 1)), 0, 1);
    }

    // ---- stage A' = g[b,e] * x[b,k], split into hi/lo bf16 planes ----
    {
      const float ge = gate[(size_t)(bM + row) * NEXP + e];
      const float4* src = (const float4*)(xin + (size_t)(bM + row) * D + ke);
#pragma unroll
      for (int i = 0; i < 4; ++i) {
        const int c4 = c4b + 4 * i;
        float4 v = src[c4];
        float vals[4] = {v.x * ge, v.y * ge, v.z * ge, v.w * ge};
#pragma unroll
        for (int j = 0; j < 4; ++j) {
          unsigned short hi, lo;
          split_f32_bf16(vals[j], hi, lo);
          sm.Ahi[row][c4 * 4 + j] = hi;
          sm.Alo[row][c4 * 4 + j] = lo;
        }
      }
    }
    // ---- stage B[n][k] = W[e, bN+n, ke+k] (row-major in k: coalesced) ----
    {
      const float4* src = (const float4*)(Wl + ((size_t)e * D + (bN + row)) * D + ke);
#pragma unroll
      for (int i = 0; i < 4; ++i) {
        const int c4 = c4b + 4 * i;
        float4 v = src[c4];
        float vals[4] = {v.x, v.y, v.z, v.w};
#pragma unroll
        for (int j = 0; j < 4; ++j) {
          unsigned short hi, lo;
          split_f32_bf16(vals[j], hi, lo);
          sm.Bhi[row][c4 * 4 + j] = hi;
          sm.Blo[row][c4 * 4 + j] = lo;
        }
      }
    }
    __syncthreads();

#pragma unroll
    for (int ks = 0; ks < BK; ks += 32) {
      Frag ah, al, bh0, bl0, bh1, bl1;
      // A 16x32 bf16 lane layout: lane = (half, M=lr); K = 16*half + [0..15]
      const int ar = m_off + lr;
      ah.q[0] = *(const uint4*)&sm.Ahi[ar][ks + 16 * half];
      ah.q[1] = *(const uint4*)&sm.Ahi[ar][ks + 16 * half + 8];
      al.q[0] = *(const uint4*)&sm.Alo[ar][ks + 16 * half];
      al.q[1] = *(const uint4*)&sm.Alo[ar][ks + 16 * half + 8];
      // B 32x16 bf16 lane layout: lane = (half, N=lr); K = 16*half + [0..15]
      const int br0 = n0 + lr, br1 = n1 + lr;
      bh0.q[0] = *(const uint4*)&sm.Bhi[br0][ks + 16 * half];
      bh0.q[1] = *(const uint4*)&sm.Bhi[br0][ks + 16 * half + 8];
      bl0.q[0] = *(const uint4*)&sm.Blo[br0][ks + 16 * half];
      bl0.q[1] = *(const uint4*)&sm.Blo[br0][ks + 16 * half + 8];
      bh1.q[0] = *(const uint4*)&sm.Bhi[br1][ks + 16 * half];
      bh1.q[1] = *(const uint4*)&sm.Bhi[br1][ks + 16 * half + 8];
      bl1.q[0] = *(const uint4*)&sm.Blo[br1][ks + 16 * half];
      bl1.q[1] = *(const uint4*)&sm.Blo[br1][ks + 16 * half + 8];

      // (ah+al)*(bh+bl) ~= ah*bh + ah*bl + al*bh  (f32-class accuracy)
      acc0 = __builtin_amdgcn_wmma_f32_16x16x32_bf16(false, ah.v, false, bh0.v, (short)0, acc0, false, false);
      acc0 = __builtin_amdgcn_wmma_f32_16x16x32_bf16(false, ah.v, false, bl0.v, (short)0, acc0, false, false);
      acc0 = __builtin_amdgcn_wmma_f32_16x16x32_bf16(false, al.v, false, bh0.v, (short)0, acc0, false, false);
      acc1 = __builtin_amdgcn_wmma_f32_16x16x32_bf16(false, ah.v, false, bh1.v, (short)0, acc1, false, false);
      acc1 = __builtin_amdgcn_wmma_f32_16x16x32_bf16(false, ah.v, false, bl1.v, (short)0, acc1, false, false);
      acc1 = __builtin_amdgcn_wmma_f32_16x16x32_bf16(false, al.v, false, bh1.v, (short)0, acc1, false, false);
    }
    __syncthreads();
  }

  // epilogue: x_out = x0*(y + sum_e g[b,e]*bias[e,o]) + x_in
  // C layout: VGPR v, lane (half,lr): M = v + 8*half, N = lr.
#pragma unroll
  for (int s = 0; s < 2; ++s) {
    const int o = bN + (s ? n1 : n0) + lr;
    v8f acc = s ? acc1 : acc0;
    float gb[8] = {0, 0, 0, 0, 0, 0, 0, 0};
#pragma unroll
    for (int e = 0; e < NEXP; ++e) {
      const float be = bl[(size_t)e * D + o];
#pragma unroll
      for (int v = 0; v < 8; ++v) {
        const int bt = bM + m_off + v + 8 * half;
        gb[v] += gate[(size_t)bt * NEXP + e] * be;
      }
    }
#pragma unroll
    for (int v = 0; v < 8; ++v) {
      const int bt = bM + m_off + v + 8 * half;
      const size_t idx = (size_t)bt * D + o;
      xout[idx] = x0[idx] * (acc[v] + gb[v]) + xin[idx];
    }
  }
}

// ---------------------------------------------------------------------------
extern "C" void kernel_launch(void* const* d_in, const int* in_sizes, int n_in,
                              void* d_out, int out_size, void* d_ws, size_t ws_size,
                              hipStream_t stream) {
  const float* x   = (const float*)d_in[0];
  const float* W   = (const float*)d_in[1];   // [L, E, D, D]
  const float* bia = (const float*)d_in[2];   // [L, E, D]
  const float* Wg1 = (const float*)d_in[3];   // [GH, D]
  const float* Wg2 = (const float*)d_in[4];   // [E, GH]
  const float* Wn  = (const float*)d_in[5];   // [E, D]
  float* out = (float*)d_out;

  // workspace: two ping-pong activation buffers + gate matrix (~16.1 MB)
  float* ws0  = (float*)d_ws;
  float* ws1  = ws0 + (size_t)BTOK * D;
  float* gate = ws1 + (size_t)BTOK * D;

  const float* cur = x;
  float* outs[LAYERS] = {ws0, ws1, ws0, out};
  for (int l = 0; l < LAYERS; ++l) {
    gate_kernel<<<BTOK / 4, 128, 0, stream>>>(cur, Wg1, Wg2, Wn, gate, l);
    moe_layer_kernel<<<dim3(D / BN, BTOK / BM), 256, 0, stream>>>(
        cur, x, W + (size_t)l * NEXP * D * D, bia + (size_t)l * NEXP * D,
        gate, outs[l]);
    cur = outs[l];
  }
  (void)in_sizes; (void)n_in; (void)out_size; (void)ws_size;
}
